// GraphEncoder_11330123727146
// MI455X (gfx1250) — matrix operand order repaired and checked by reading, not compiled, not run on hardware
//
#include <hip/hip_runtime.h>
#include <hip/hip_bf16.h>

// ---------------------------------------------------------------------------
// GNN encoder for MI455X (gfx1250): per-edge MLP via v_wmma_f32_16x16x32_f16,
// scatter-mean via global f32 atomics (L2-resident accumulators).
// ---------------------------------------------------------------------------

typedef __attribute__((ext_vector_type(16))) _Float16 v16h;
typedef __attribute__((ext_vector_type(8)))  float    v8f;

union HFrag { v16h v; uint4 q[2]; };

// A-fragment (16x32 f16, M x K). Per ISA: lane L (row = L&15, hi = L>>4):
//   elements 0..7  = K[hi*8      .. hi*8+7]
//   elements 8..15 = K[16 + hi*8 .. 16+hi*8+7]
__device__ __forceinline__ v16h ldA(const _Float16* row, int k0, int hi) {
  HFrag f;
  f.q[0] = *(const uint4*)(row + k0 + hi * 8);
  f.q[1] = *(const uint4*)(row + k0 + 16 + hi * 8);
  return f.v;
}

// B-fragment (32x16 f16, K x N) from transposed weights Wt[n][k]:
// lane L (col = L&15, hi = L>>4): elements 0..7 = K[hi*8 .. hi*8+7],
// elements 8..15 = K[16 + hi*8 .. 16+hi*8+7] (same order as ldA).
__device__ __forceinline__ v16h ldB(const _Float16* wrow, int k0, int hi) {
  HFrag f;
  f.q[0] = *(const uint4*)(wrow + k0 + hi * 8);
  f.q[1] = *(const uint4*)(wrow + k0 + 16 + hi * 8);
  return f.v;
}

// ----------------------------- prep kernels --------------------------------

__global__ __launch_bounds__(256) void cvt_f32_to_f16_kernel(
    const float* __restrict__ in, _Float16* __restrict__ out, int n) {
  for (int i = blockIdx.x * blockDim.x + threadIdx.x; i < n;
       i += gridDim.x * blockDim.x)
    out[i] = (_Float16)in[i];
}

// W is [K][N] row-major (jax layout); produce Wt[n][k] f16.
__global__ __launch_bounds__(256) void transpose_w_kernel(
    const float* __restrict__ W, _Float16* __restrict__ Wt, int K, int N) {
  int i = blockIdx.x * blockDim.x + threadIdx.x;
  if (i < K * N) {
    int k = i / N, n = i % N;
    Wt[(size_t)n * K + k] = (_Float16)W[i];
  }
}

__global__ __launch_bounds__(256) void count_kernel(
    const long long* __restrict__ dst, float* __restrict__ cnt, int E) {
  for (int e = blockIdx.x * blockDim.x + threadIdx.x; e < E;
       e += gridDim.x * blockDim.x)
    atomicAdd(cnt + (int)dst[e], 1.0f);
}

// mean + relu + f16 recode for the inter-layer node features
__global__ __launch_bounds__(256) void node_act_kernel(
    const float* __restrict__ agg, const float* __restrict__ cnt,
    _Float16* __restrict__ h16, int N, int C) {
  int i = blockIdx.x * blockDim.x + threadIdx.x;
  if (i < N * C) {
    int n = i / C;
    float d = cnt[n]; d = d > 1.0f ? d : 1.0f;
    float v = agg[i] / d;
    v = v > 0.0f ? v : 0.0f;
    h16[i] = (_Float16)v;
  }
}

__global__ __launch_bounds__(256) void final_mean_kernel(
    const float* __restrict__ agg, const float* __restrict__ cnt,
    float* __restrict__ out, int N, int C) {
  int i = blockIdx.x * blockDim.x + threadIdx.x;
  if (i < N * C) {
    int n = i / C;
    float d = cnt[n]; d = d > 1.0f ? d : 1.0f;
    out[i] = agg[i] / d;
  }
}

// ------------------------- fused per-edge MLP ------------------------------
// One block (8 wave32s) handles a tile of 16 edges end-to-end:
//   gather cat(feat[dst],feat[src]) -> LDS A[16][DIN]
//   GEMM1 (WMMA) + bias + relu -> LDS Ht[16][HID] f16
//   GEMM2 (WMMA) + bias -> atomicAdd into agg[dst]
template <int DIN, int HID, int DOUT>
__global__ __launch_bounds__(256) void edge_mlp_kernel(
    const _Float16* __restrict__ feat,   // [N][DIN/2] f16 node features
    const long long* __restrict__ src,   // [E]
    const long long* __restrict__ dst,   // [E]
    const _Float16* __restrict__ WaT,    // [HID][DIN]
    const float*    __restrict__ ba,     // [HID]
    const _Float16* __restrict__ WbT,    // [DOUT][HID]
    const float*    __restrict__ bb,     // [DOUT]
    float* __restrict__ agg,             // [N][DOUT]
    int E) {
  static_assert(DIN % 32 == 0 && HID % 128 == 0 && DOUT % 128 == 0, "dims");
  constexpr int NODE_H = DIN / 2;     // halves per node feature row
  constexpr int CH     = DIN / 16;    // halves gathered per thread

  __shared__ _Float16 A[16 * DIN];
  __shared__ _Float16 Ht[16 * HID];
  __shared__ int sDst[16];
  __shared__ int sSrc[16];

  const int tid = threadIdx.x;
  const long long e0 = (long long)blockIdx.x * 16;

  if (tid < 16) {
    long long e = e0 + tid;
    if (e < E) {
      sDst[tid] = (int)dst[e];
      sSrc[tid] = (int)src[e];
    } else {
      sDst[tid] = -1;
      sSrc[tid] = -1;
    }
  }
  __syncthreads();

  // gather cat(feat[dst], feat[src]) into LDS, 32B/64B chunks per thread
  {
    int r = tid >> 4, seg = tid & 15;
    int off = seg * CH;
    int node, loc;
    if (off < NODE_H) { node = sDst[r]; loc = off; }
    else              { node = sSrc[r]; loc = off - NODE_H; }
    if (node < 0) node = 0;  // tail guard (E % 16 == 0 in practice)
    const uint4* gp = (const uint4*)(feat + (size_t)node * NODE_H + loc);
    uint4* lp = (uint4*)(A + r * DIN + off);
#pragma unroll
    for (int i = 0; i < CH / 8; ++i) lp[i] = gp[i];
  }
  __syncthreads();

  const int wave = tid >> 5;
  const int lane = tid & 31;
  const int hi   = lane >> 4;
  const int lrow = lane & 15;

  // ---- GEMM1: [16 x DIN] @ [DIN x HID], each wave owns HID/8 columns ----
  constexpr int NT1 = HID / 128;  // n-tiles of 16 per wave
#pragma unroll
  for (int t = 0; t < NT1; ++t) {
    const int n0 = (wave * NT1 + t) * 16;
    v8f acc = {};
    const _Float16* arow = A + lrow * DIN;
    const _Float16* brow = WaT + (size_t)(n0 + lrow) * DIN;
#pragma unroll
    for (int k0 = 0; k0 < DIN; k0 += 32) {
      v16h a = ldA(arow, k0, hi);
      v16h b = ldB(brow, k0, hi);
      acc = __builtin_amdgcn_wmma_f32_16x16x32_f16(
          false, a, false, b, (short)0, acc, false, false);
    }
    const float bias = ba[n0 + lrow];
#pragma unroll
    for (int r = 0; r < 8; ++r) {
      float v = acc[r] + bias;          // C elem r: row = hi*8+r, col = lrow
      v = v > 0.0f ? v : 0.0f;
      Ht[(hi * 8 + r) * HID + n0 + lrow] = (_Float16)v;
    }
  }
  __syncthreads();

  // ---- GEMM2: [16 x HID] @ [HID x DOUT] + scatter-add into agg[dst] ----
  constexpr int NT2 = DOUT / 128;
#pragma unroll
  for (int t = 0; t < NT2; ++t) {
    const int n0 = (wave * NT2 + t) * 16;
    v8f acc = {};
    const _Float16* arow = Ht + lrow * HID;
    const _Float16* brow = WbT + (size_t)(n0 + lrow) * HID;
#pragma unroll
    for (int k0 = 0; k0 < HID; k0 += 32) {
      v16h a = ldA(arow, k0, hi);
      v16h b = ldB(brow, k0, hi);
      acc = __builtin_amdgcn_wmma_f32_16x16x32_f16(
          false, a, false, b, (short)0, acc, false, false);
    }
    const float bias = bb[n0 + lrow];
#pragma unroll
    for (int r = 0; r < 8; ++r) {
      int node = sDst[hi * 8 + r];
      if (node >= 0)
        atomicAdd(agg + (size_t)node * DOUT + n0 + lrow, acc[r] + bias);
    }
  }
}

// ------------------------------ launcher -----------------------------------

extern "C" void kernel_launch(void* const* d_in, const int* in_sizes, int n_in,
                              void* d_out, int out_size, void* d_ws, size_t ws_size,
                              hipStream_t stream) {
  constexpr int ND = 128, HD = 256, OD = 128;
  const int N = in_sizes[0] / ND;        // 20000
  const int E = in_sizes[1] / 2;         // 640000

  const float*     x   = (const float*)d_in[0];
  const long long* ei  = (const long long*)d_in[1];  // int64 (2, E)
  const long long* src = ei;
  const long long* dstI = ei + E;
  const float* W1a = (const float*)d_in[2];
  const float* b1a = (const float*)d_in[3];
  const float* W1b = (const float*)d_in[4];
  const float* b1b = (const float*)d_in[5];
  const float* W2a = (const float*)d_in[6];
  const float* b2a = (const float*)d_in[7];
  const float* W2b = (const float*)d_in[8];
  const float* b2b = (const float*)d_in[9];
  float* out = (float*)d_out;

  // workspace carve-up (~47 MB)
  char* p = (char*)d_ws;
  _Float16* x16  = (_Float16*)p; p += (size_t)N * ND * 2;          // 5.12 MB
  _Float16* W1aT = (_Float16*)p; p += (size_t)(2*ND) * HD * 2;     // 128 KB
  _Float16* W1bT = (_Float16*)p; p += (size_t)HD * HD * 2;         // 128 KB
  _Float16* W2aT = (_Float16*)p; p += (size_t)(2*HD) * OD * 2;     // 128 KB
  _Float16* W2bT = (_Float16*)p; p += (size_t)OD * OD * 2;         // 32 KB
  float*    cnt  = (float*)p;    p += (size_t)N * 4;               // 80 KB
  float*    agg1 = (float*)p;    p += (size_t)N * HD * 4;          // 20.5 MB
  _Float16* h16  = (_Float16*)p; p += (size_t)N * HD * 2;          // 10.2 MB
  float*    agg2 = (float*)p;    p += (size_t)N * OD * 4;          // 10.2 MB

  // zero accumulators & counts (graph-capture safe)
  hipMemsetAsync(cnt,  0, (size_t)N * 4, stream);
  hipMemsetAsync(agg1, 0, (size_t)N * HD * 4, stream);
  hipMemsetAsync(agg2, 0, (size_t)N * OD * 4, stream);

  // f16 conversions + weight transposes
  {
    int n = N * ND;
    cvt_f32_to_f16_kernel<<<(n + 255) / 256, 256, 0, stream>>>(x, x16, n);
  }
  transpose_w_kernel<<<((2*ND)*HD + 255) / 256, 256, 0, stream>>>(W1a, W1aT, 2*ND, HD);
  transpose_w_kernel<<<(HD*HD + 255) / 256, 256, 0, stream>>>(W1b, W1bT, HD, HD);
  transpose_w_kernel<<<((2*HD)*OD + 255) / 256, 256, 0, stream>>>(W2a, W2aT, 2*HD, OD);
  transpose_w_kernel<<<(OD*OD + 255) / 256, 256, 0, stream>>>(W2b, W2bT, OD, OD);

  // degree counts
  count_kernel<<<1024, 256, 0, stream>>>(dstI, cnt, E);

  const int tiles = (E + 15) / 16;

  // layer 1: DIN=256, HID=256, DOUT=256
  edge_mlp_kernel<256, 256, 256><<<tiles, 256, 0, stream>>>(
      x16, src, dstI, W1aT, b1a, W1bT, b1b, agg1, E);

  // inter-layer: mean + relu + f16 recode
  node_act_kernel<<<(N * HD + 255) / 256, 256, 0, stream>>>(agg1, cnt, h16, N, HD);

  // layer 2: DIN=512, HID=128, DOUT=128
  edge_mlp_kernel<512, 128, 128><<<tiles, 256, 0, stream>>>(
      h16, src, dstI, W2aT, b2a, W2bT, b2b, agg2, E);

  // final scatter-mean
  final_mean_kernel<<<(N * OD + 255) / 256, 256, 0, stream>>>(agg2, cnt, out, N, OD);
}
